// MetaNet_43757126811844
// MI455X (gfx1250) — matrix operand order repaired and checked by reading, not compiled, not run on hardware
//
#include <hip/hip_runtime.h>

typedef __attribute__((ext_vector_type(16))) _Float16 v16h;
typedef __attribute__((ext_vector_type(8)))  _Float16 v8h;
typedef __attribute__((ext_vector_type(8)))  float    v8f;

// ---------------------------------------------------------------------------
// WMMA fragment loaders (CDNA5 16x16x32 f16, wave32).
//
// A (16x32, MxK): lane L holds row m = L%16; half = L/16.
//   v16h elems 0..7  = K[k0 + half*16 .. +7]
//   v16h elems 8..15 = K[k0 + half*16 + 8 .. +7]
// B (32x16, KxN): lane L holds col n = L%16; half = L/16.
//   v16h elems j = W[k0 + half*16 + j][n]   (dense analog of ISA sparse-B)
// C/D (16x16 f32): lane L col n = L%16; vgpr r holds row m = 8*(L/16) + r.
// ---------------------------------------------------------------------------
__device__ __forceinline__ v16h ldA(const _Float16* A, int lda, int mBase,
                                    int k0, int lane) {
  const int m    = mBase + (lane & 15);
  const int half = lane >> 4;
  const _Float16* p = A + m * lda + k0 + half * 16;
  v8h lo = *(const v8h*)p;
  v8h hi = *(const v8h*)(p + 8);
  v16h a;
#pragma unroll
  for (int j = 0; j < 8; ++j) { a[j] = lo[j]; a[8 + j] = hi[j]; }
  return a;
}

// wbuf holds one pre-swizzled 32x64 weight chunk:
//   wbuf[(n*2 + half)*16 + j] = W[k0 + half*16 + j][n]  -> contiguous 32B/frag
__device__ __forceinline__ v16h ldB(const _Float16* wbuf, int n, int half) {
  const _Float16* p = wbuf + (n * 2 + half) * 16;
  v8h lo = *(const v8h*)p;
  v8h hi = *(const v8h*)(p + 8);
  v16h b;
#pragma unroll
  for (int j = 0; j < 8; ++j) { b[j] = lo[j]; b[8 + j] = hi[j]; }
  return b;
}

// C(64x64) += A(64xK) @ W(Kx64). 4 waves: wave w owns rows [16w,16w+16),
// loops the four 16-col tiles. Weights streamed from global (pre-swizzled
// f16) through a 32x64 LDS chunk (4KB) per K-step.
__device__ __forceinline__ void gemm_accum(const _Float16* A, int lda, int K,
                                           const _Float16* Wsw, _Float16* wbuf,
                                           int tid, v8f acc[4]) {
  const int lane  = tid & 31;
  const int mBase = (tid >> 5) << 4;
  const v8f z = {};
#pragma unroll
  for (int nt = 0; nt < 4; ++nt) acc[nt] = z;
  for (int k0 = 0; k0 < K; k0 += 32) {
    __syncthreads();  // previous consumers of wbuf / producers of A done
    const _Float16* src = Wsw + (size_t)(k0 >> 5) * 2048;
    *(v8h*)(wbuf + tid * 8)        = *(const v8h*)(src + tid * 8);
    *(v8h*)(wbuf + 1024 + tid * 8) = *(const v8h*)(src + 1024 + tid * 8);
    __syncthreads();
    v16h a = ldA(A, lda, mBase, k0, lane);
#pragma unroll
    for (int nt = 0; nt < 4; ++nt) {
      v16h b = ldB(wbuf, (nt << 4) + (lane & 15), lane >> 4);
      acc[nt] = __builtin_amdgcn_wmma_f32_16x16x32_f16(
          false, a, false, b, (short)0, acc[nt], false, false);
    }
  }
}

// ---- epilogues (fused bias / relu) ----------------------------------------
__device__ __forceinline__ void epi_to_lds(const v8f acc[4], const float* biasG,
                                           bool relu, _Float16* outL, int ldo,
                                           int colOff, int tid) {
  const int lane = tid & 31, half = lane >> 4, n0 = lane & 15;
  const int mBase = (tid >> 5) << 4;
#pragma unroll
  for (int nt = 0; nt < 4; ++nt) {
    const int n = (nt << 4) + n0;
    const float bv = biasG[n];
#pragma unroll
    for (int r = 0; r < 8; ++r) {
      float v = acc[nt][r] + bv;
      if (relu) v = fmaxf(v, 0.0f);
      const int m = mBase + half * 8 + r;
      outL[m * ldo + colOff + n] = (_Float16)v;
    }
  }
}

__device__ __forceinline__ void epi_ea(const v8f acc[4], const float* biasG,
                                       _Float16* outL, int ldo, int colOff,
                                       _Float16* outG, long long e0,
                                       long long E, int tid) {
  const int lane = tid & 31, half = lane >> 4, n0 = lane & 15;
  const int mBase = (tid >> 5) << 4;
#pragma unroll
  for (int nt = 0; nt < 4; ++nt) {
    const int n = (nt << 4) + n0;
    const float bv = biasG[n];
#pragma unroll
    for (int r = 0; r < 8; ++r) {
      const int m = mBase + half * 8 + r;
      const _Float16 hv = (_Float16)(acc[nt][r] + bv);
      outL[m * ldo + colOff + n] = hv;
      if (e0 + m < E) outG[(size_t)(e0 + m) * 64 + n] = hv;
    }
  }
}

__device__ __forceinline__ void epi_scatter(const v8f acc[4], const float* biasG,
                                            float* agg, const int* rowsL,
                                            long long e0, long long E, int tid) {
  const int lane = tid & 31, half = lane >> 4, n0 = lane & 15;
  const int mBase = (tid >> 5) << 4;
#pragma unroll
  for (int nt = 0; nt < 4; ++nt) {
    const int n = (nt << 4) + n0;
    const float bv = biasG[n];
#pragma unroll
    for (int r = 0; r < 8; ++r) {
      const int m = mBase + half * 8 + r;
      if (e0 + m < E)
        atomicAdd(agg + (size_t)rowsL[m] * 64 + n, acc[nt][r] + bv);
    }
  }
}

__device__ __forceinline__ void epi_node_x1(const v8f acc[4], const float* biasG,
                                            _Float16* x1g, long long v0,
                                            long long N, int tid) {
  const int lane = tid & 31, half = lane >> 4, n0 = lane & 15;
  const int mBase = (tid >> 5) << 4;
#pragma unroll
  for (int nt = 0; nt < 4; ++nt) {
    const int n = (nt << 4) + n0;
    const float bv = biasG[n];
#pragma unroll
    for (int r = 0; r < 8; ++r) {
      const int m = mBase + half * 8 + r;
      const long long v = v0 + m;
      if (v < N)
        x1g[(size_t)v * 64 + n] = (_Float16)fmaxf(acc[nt][r] + bv, 0.0f);
    }
  }
}

// ---------------------------------------------------------------------------
// Layer 1, edge side: EdgeModel (96->64->64) + NodeModel-a (96->64->64),
// scatter-add to agg1/cnt1, emit ea1 (f16).
// ---------------------------------------------------------------------------
__global__ __launch_bounds__(128) void edge_layer1(
    const int* __restrict__ ei, const float* __restrict__ eattr,
    const _Float16* __restrict__ x16,
    const _Float16* w_e1, const float* b_e1,
    const _Float16* w_e2, const float* b_e2,
    const _Float16* w_a1, const float* b_a1,
    const _Float16* w_a2, const float* b_a2,
    _Float16* __restrict__ ea1, float* __restrict__ agg1,
    float* __restrict__ cnt1, long long E) {
  __shared__ __align__(16) _Float16 A1[64 * 96];   // [x_row | x_col | eattr]
  __shared__ __align__(16) _Float16 A2[64 * 96];   // [x_col | ea]
  __shared__ __align__(16) _Float16 Ht[64 * 64];
  __shared__ __align__(16) _Float16 wbuf[32 * 64];
  __shared__ int rows[64], cols[64];
  const int tid = threadIdx.x;
  const long long e0 = (long long)blockIdx.x * 64;

  if (tid < 64) {
    long long e = e0 + tid; if (e >= E) e = E - 1;
    rows[tid] = ei[e];
    cols[tid] = ei[E + e];
    if (e0 + tid < E) atomicAdd(cnt1 + rows[tid], 1.0f);
  }
  __syncthreads();
  for (int i = tid; i < 64 * 32; i += 128) {
    const int m = i >> 5, f = i & 31;
    const _Float16 xr = x16[(size_t)rows[m] * 32 + f];
    const _Float16 xc = x16[(size_t)cols[m] * 32 + f];
    A1[m * 96 + f] = xr;
    A1[m * 96 + 32 + f] = xc;
    A2[m * 96 + f] = xc;
    long long e = e0 + m; if (e >= E) e = E - 1;
    A1[m * 96 + 64 + f] = (_Float16)eattr[(size_t)e * 32 + f];
  }

  v8f acc[4];
  gemm_accum(A1, 96, 96, w_e1, wbuf, tid, acc);
  epi_to_lds(acc, b_e1, true, Ht, 64, 0, tid);
  gemm_accum(Ht, 64, 64, w_e2, wbuf, tid, acc);
  epi_ea(acc, b_e2, A2, 96, 32, ea1, e0, E, tid);      // ea -> A2[:,32:96] + HBM
  gemm_accum(A2, 96, 96, w_a1, wbuf, tid, acc);
  epi_to_lds(acc, b_a1, true, Ht, 64, 0, tid);
  gemm_accum(Ht, 64, 64, w_a2, wbuf, tid, acc);
  epi_scatter(acc, b_a2, agg1, rows, e0, E, tid);
}

// Layer 1, node side: NodeModel-b (96->64->64) + outer relu -> x1 (f16).
__global__ __launch_bounds__(128) void node_layer1(
    const _Float16* __restrict__ x16, const float* __restrict__ agg1,
    const float* __restrict__ cnt1,
    const _Float16* w1, const float* b1, const _Float16* w2, const float* b2,
    _Float16* __restrict__ x1, long long N) {
  __shared__ __align__(16) _Float16 A[64 * 96];      // [x | mean_agg]
  __shared__ __align__(16) _Float16 Ht[64 * 64];
  __shared__ __align__(16) _Float16 wbuf[32 * 64];
  const int tid = threadIdx.x;
  const long long v0 = (long long)blockIdx.x * 64;
  for (int i = tid; i < 64 * 32; i += 128) {
    const int m = i >> 5, f = i & 31;
    long long v = v0 + m; if (v >= N) v = N - 1;
    A[m * 96 + f] = x16[(size_t)v * 32 + f];
  }
  for (int i = tid; i < 64 * 64; i += 128) {
    const int m = i >> 6, f = i & 63;
    long long v = v0 + m; if (v >= N) v = N - 1;
    const float c = fmaxf(cnt1[v], 1.0f);
    A[m * 96 + 32 + f] = (_Float16)(agg1[(size_t)v * 64 + f] / c);
  }
  v8f acc[4];
  gemm_accum(A, 96, 96, w1, wbuf, tid, acc);
  epi_to_lds(acc, b1, true, Ht, 64, 0, tid);
  gemm_accum(Ht, 64, 64, w2, wbuf, tid, acc);
  epi_node_x1(acc, b2, x1, v0, N, tid);               // relu(x1) fused
}

// Layer 2, edge side: EdgeModel (192->64->64) + NodeModel-a (128->64->64).
__global__ __launch_bounds__(128) void edge_layer2(
    const int* __restrict__ ei, const _Float16* __restrict__ x1,
    const _Float16* __restrict__ ea1,
    const _Float16* w_e1, const float* b_e1,
    const _Float16* w_e2, const float* b_e2,
    const _Float16* w_a1, const float* b_a1,
    const _Float16* w_a2, const float* b_a2,
    float* __restrict__ agg2, float* __restrict__ cnt2, long long E) {
  __shared__ __align__(16) _Float16 A1[64 * 192];    // [x1_row | x1_col | ea1]
  __shared__ __align__(16) _Float16 A2[64 * 128];    // [x1_col | ea2]
  __shared__ __align__(16) _Float16 Ht[64 * 64];
  __shared__ __align__(16) _Float16 wbuf[32 * 64];
  __shared__ int rows[64], cols[64];
  const int tid = threadIdx.x;
  const long long e0 = (long long)blockIdx.x * 64;

  if (tid < 64) {
    long long e = e0 + tid; if (e >= E) e = E - 1;
    rows[tid] = ei[e];
    cols[tid] = ei[E + e];
    if (e0 + tid < E) atomicAdd(cnt2 + rows[tid], 1.0f);
  }
  __syncthreads();
  for (int i = tid; i < 64 * 64; i += 128) {
    const int m = i >> 6, f = i & 63;
    const _Float16 xr = x1[(size_t)rows[m] * 64 + f];
    const _Float16 xc = x1[(size_t)cols[m] * 64 + f];
    A1[m * 192 + f] = xr;
    A1[m * 192 + 64 + f] = xc;
    A2[m * 128 + f] = xc;
    long long e = e0 + m; if (e >= E) e = E - 1;
    A1[m * 192 + 128 + f] = ea1[(size_t)e * 64 + f];
  }

  v8f acc[4];
  gemm_accum(A1, 192, 192, w_e1, wbuf, tid, acc);
  epi_to_lds(acc, b_e1, true, Ht, 64, 0, tid);
  gemm_accum(Ht, 64, 64, w_e2, wbuf, tid, acc);
  epi_to_lds(acc, b_e2, false, A2, 128, 64, tid);     // ea2 kept on-chip only
  gemm_accum(A2, 128, 128, w_a1, wbuf, tid, acc);
  epi_to_lds(acc, b_a1, true, Ht, 64, 0, tid);
  gemm_accum(Ht, 64, 64, w_a2, wbuf, tid, acc);
  epi_scatter(acc, b_a2, agg2, rows, e0, E, tid);
}

// Layer 2, node side: NodeModel-b (128->64->1) -> d_out (f32).
__global__ __launch_bounds__(128) void node_layer2(
    const _Float16* __restrict__ x1, const float* __restrict__ agg2,
    const float* __restrict__ cnt2,
    const _Float16* w1, const float* b1,
    const _Float16* __restrict__ w2h, const float* __restrict__ b2,
    float* __restrict__ out, long long N) {
  __shared__ __align__(16) _Float16 A[64 * 128];     // [x1 | mean_agg2]
  __shared__ __align__(16) _Float16 Ht[64 * 64];
  __shared__ __align__(16) _Float16 wbuf[32 * 64];
  const int tid = threadIdx.x;
  const long long v0 = (long long)blockIdx.x * 64;
  for (int i = tid; i < 64 * 64; i += 128) {
    const int m = i >> 6, f = i & 63;
    long long v = v0 + m; if (v >= N) v = N - 1;
    A[m * 128 + f] = x1[(size_t)v * 64 + f];
    const float c = fmaxf(cnt2[v], 1.0f);
    A[m * 128 + 64 + f] = (_Float16)(agg2[(size_t)v * 64 + f] / c);
  }
  v8f acc[4];
  gemm_accum(A, 128, 128, w1, wbuf, tid, acc);
  epi_to_lds(acc, b1, true, Ht, 64, 0, tid);
  __syncthreads();
  if (tid < 64) {                                    // final 64 -> 1: VALU dot
    const long long v = v0 + tid;
    if (v < N) {
      float s = b2[0];
#pragma unroll
      for (int k = 0; k < 64; ++k)
        s += (float)Ht[tid * 64 + k] * (float)w2h[k];
      out[v] = s;
    }
  }
}

// ---------------------------------------------------------------------------
// Prep kernels.
// ---------------------------------------------------------------------------
__global__ void k_zero(float* p, long long n) {
  const long long i = (long long)blockIdx.x * blockDim.x + threadIdx.x;
  if (i < n) p[i] = 0.0f;
}
__global__ void k_cvt(const float* s, _Float16* d, long long n) {
  const long long i = (long long)blockIdx.x * blockDim.x + threadIdx.x;
  if (i < n) d[i] = (_Float16)s[i];
}
// f32 W[K][64] row-major -> f16 WMMA-B swizzle:
// sw[c*2048 + n*32 + half*16 + j] = W[(c*32 + half*16 + j)*64 + n]
__global__ void k_cvt_swz(const float* W, _Float16* sw, int K) {
  const int idx = blockIdx.x * blockDim.x + threadIdx.x;
  if (idx >= K * 64) return;
  const int c = idx >> 11;
  const int r = idx & 2047;
  const int n = r >> 5;
  const int h = (r >> 4) & 1;
  const int j = r & 15;
  sw[idx] = (_Float16)W[((c << 5) + (h << 4) + j) * 64 + n];
}

// ---------------------------------------------------------------------------
extern "C" void kernel_launch(void* const* d_in, const int* in_sizes, int n_in,
                              void* d_out, int out_size, void* d_ws,
                              size_t ws_size, hipStream_t stream) {
  (void)in_sizes; (void)n_in; (void)out_size; (void)ws_size;
  const long long N = 100000, E = 1600000;

  const float* x     = (const float*)d_in[0];
  const float* eattr = (const float*)d_in[1];
  const int*   ei    = (const int*)d_in[2];
  const float* e1w1 = (const float*)d_in[3];  const float* e1b1 = (const float*)d_in[4];
  const float* e1w2 = (const float*)d_in[5];  const float* e1b2 = (const float*)d_in[6];
  const float* n1aw1 = (const float*)d_in[7]; const float* n1ab1 = (const float*)d_in[8];
  const float* n1aw2 = (const float*)d_in[9]; const float* n1ab2 = (const float*)d_in[10];
  const float* n1bw1 = (const float*)d_in[11]; const float* n1bb1 = (const float*)d_in[12];
  const float* n1bw2 = (const float*)d_in[13]; const float* n1bb2 = (const float*)d_in[14];
  const float* e2w1 = (const float*)d_in[15]; const float* e2b1 = (const float*)d_in[16];
  const float* e2w2 = (const float*)d_in[17]; const float* e2b2 = (const float*)d_in[18];
  const float* n2aw1 = (const float*)d_in[19]; const float* n2ab1 = (const float*)d_in[20];
  const float* n2aw2 = (const float*)d_in[21]; const float* n2ab2 = (const float*)d_in[22];
  const float* n2bw1 = (const float*)d_in[23]; const float* n2bb1 = (const float*)d_in[24];
  const float* n2bw2 = (const float*)d_in[25]; const float* n2bb2 = (const float*)d_in[26];

  char* ws = (char*)d_ws;
  size_t off = 0;
  auto take = [&](size_t bytes) -> void* {
    void* p = ws + off;
    off = (off + bytes + 255) & ~(size_t)255;
    return p;
  };

  float* agg1 = (float*)take((size_t)N * 64 * 4);
  float* cnt1 = (float*)take((size_t)N * 4);
  float* agg2 = (float*)take((size_t)N * 64 * 4);
  float* cnt2 = (float*)take((size_t)N * 4);
  const long long zero_floats = (long long)(off / 4);  // zero agg1..cnt2 (+pad)
  _Float16* x16  = (_Float16*)take((size_t)N * 32 * 2);
  _Float16* x1f  = (_Float16*)take((size_t)N * 64 * 2);
  _Float16* ea1f = (_Float16*)take((size_t)E * 64 * 2);
  _Float16* sE1W1  = (_Float16*)take(96 * 64 * 2);
  _Float16* sE1W2  = (_Float16*)take(64 * 64 * 2);
  _Float16* sN1AW1 = (_Float16*)take(96 * 64 * 2);
  _Float16* sN1AW2 = (_Float16*)take(64 * 64 * 2);
  _Float16* sN1BW1 = (_Float16*)take(96 * 64 * 2);
  _Float16* sN1BW2 = (_Float16*)take(64 * 64 * 2);
  _Float16* sE2W1  = (_Float16*)take(192 * 64 * 2);
  _Float16* sE2W2  = (_Float16*)take(64 * 64 * 2);
  _Float16* sN2AW1 = (_Float16*)take(128 * 64 * 2);
  _Float16* sN2AW2 = (_Float16*)take(64 * 64 * 2);
  _Float16* sN2BW1 = (_Float16*)take(128 * 64 * 2);
  _Float16* w2h    = (_Float16*)take(64 * 2);

  // prep: zero accumulators, cast x to f16, swizzle all GEMM weights to f16
  k_zero<<<(int)((zero_floats + 255) / 256), 256, 0, stream>>>(agg1, zero_floats);
  k_cvt<<<(int)((N * 32 + 255) / 256), 256, 0, stream>>>(x, x16, N * 32);
  auto swz = [&](const float* W, _Float16* sw, int K) {
    k_cvt_swz<<<(K * 64 + 255) / 256, 256, 0, stream>>>(W, sw, K);
  };
  swz(e1w1, sE1W1, 96);   swz(e1w2, sE1W2, 64);
  swz(n1aw1, sN1AW1, 96); swz(n1aw2, sN1AW2, 64);
  swz(n1bw1, sN1BW1, 96); swz(n1bw2, sN1BW2, 64);
  swz(e2w1, sE2W1, 192);  swz(e2w2, sE2W2, 64);
  swz(n2aw1, sN2AW1, 128); swz(n2aw2, sN2AW2, 64);
  swz(n2bw1, sN2BW1, 128);
  k_cvt<<<1, 64, 0, stream>>>(n2bw2, w2h, 64);

  const int ebl = (int)((E + 63) / 64);
  const int nbl = (int)((N + 63) / 64);
  edge_layer1<<<ebl, 128, 0, stream>>>(ei, eattr, x16,
      sE1W1, e1b1, sE1W2, e1b2, sN1AW1, n1ab1, sN1AW2, n1ab2,
      ea1f, agg1, cnt1, E);
  node_layer1<<<nbl, 128, 0, stream>>>(x16, agg1, cnt1,
      sN1BW1, n1bb1, sN1BW2, n1bb2, x1f, N);
  edge_layer2<<<ebl, 128, 0, stream>>>(ei, x1f, ea1f,
      sE2W1, e2b1, sE2W2, e2b2, sN2AW1, n2ab1, sN2AW2, n2ab2,
      agg2, cnt2, E);
  node_layer2<<<nbl, 128, 0, stream>>>(x1f, agg2, cnt2,
      sN2BW1, n2bb1, w2h, n2bb2, (float*)d_out, N);
}
